// SwinTransformerBlock_13108240187413
// MI455X (gfx1250) — matrix operand order repaired and checked by reading, not compiled, not run on hardware
//
#include <hip/hip_runtime.h>
#include <hip/hip_bf16.h>

typedef _Float16 fp16_t;
typedef __attribute__((ext_vector_type(16))) _Float16 v16h;
typedef __attribute__((ext_vector_type(8)))  float    v8f;

#define DIM   256
#define HEADS 8
#define HD    32
#define NTOK  49
#define NPAD  64
#define HID   1024
#define IMG   112
#define NWSIDE 16     // 112/7 windows per side

#define XS 264        // LDS stride for 256-wide f16 rows (pad 8)
#define QS 776        // LDS stride for 768-wide qkv rows
#define PS 72         // LDS stride for 64-wide prob rows

__device__ __forceinline__ v8f zero8() {
  v8f c;
#pragma unroll
  for (int i = 0; i < 8; ++i) c[i] = 0.f;
  return c;
}

__device__ __forceinline__ v8f wmma32(v16h a, v16h b, v8f c) {
  // D = A(16x32 f16) * B(32x16 f16) + C(16x16 f32)
  return __builtin_amdgcn_wmma_f32_16x16x32_f16(false, a, false, b, (short)0, c,
                                                false, false);
}

// A fragment 16x32 f16: element (m,k) at p[(m0+m)*stride + k0+k]
// Contiguous k order used here: lanes 0-15 -> K 0..15; lanes 16-31 -> K 16..31
__device__ __forceinline__ v16h load_A(const fp16_t* p, int stride, int m0, int k0) {
  const int lane = threadIdx.x & 31;
  const fp16_t* r = p + (m0 + (lane & 15)) * stride + k0 + ((lane >> 4) << 4);
  v16h a;
#pragma unroll
  for (int v = 0; v < 4; ++v) {
    a[2 * v]         = r[2 * v];
    a[2 * v + 1]     = r[2 * v + 1];
    a[8 + 2 * v]     = r[8 + 2 * v];
    a[8 + 2 * v + 1] = r[8 + 2 * v + 1];
  }
  return a;
}

// B fragment 32x16 f16, K-contiguous storage: element (k,n) at p[(n0+n)*stride + k0+k]
// Contiguous k order used here: lanes 0-15 col=lane K 0..15 ; lanes 16-31 col=lane-16 K 16..31
__device__ __forceinline__ v16h load_B_kc(const fp16_t* p, int stride, int k0, int n0) {
  const int lane = threadIdx.x & 31;
  const fp16_t* r = p + (n0 + (lane & 15)) * stride + k0 + ((lane >> 4) << 4);
  v16h b;
#pragma unroll
  for (int v = 0; v < 8; ++v) { b[2 * v] = r[2 * v]; b[2 * v + 1] = r[2 * v + 1]; }
  return b;
}

// B fragment 32x16 f16, row-major storage: element (k,n) at p[(k0+k)*stride + n0+n]
__device__ __forceinline__ v16h load_B_rm(const fp16_t* p, int stride, int k0, int n0) {
  const int lane = threadIdx.x & 31;
  const int n  = n0 + (lane & 15);
  const int kb = k0 + ((lane >> 4) << 4);
  v16h b;
#pragma unroll
  for (int v = 0; v < 8; ++v) {
    b[2 * v]     = p[(kb + 2 * v) * stride + n];
    b[2 * v + 1] = p[(kb + 2 * v + 1) * stride + n];
  }
  return b;
}

// ---------------- prep kernels ----------------

// dst[n*K + k] = (f16) src[k*N + n]  (K = input dim, N = output dim)
__global__ void k_transpose_f16(const float* __restrict__ src, fp16_t* __restrict__ dst,
                                int K, int N) {
  int i = blockIdx.x * blockDim.x + threadIdx.x;
  if (i >= K * N) return;
  int n = i / K, k = i % K;
  dst[(size_t)n * K + k] = (fp16_t)src[(size_t)k * N + n];
}

// rpb[h][64][64], zero padded past 49
__global__ void k_rpb(const float* __restrict__ table, float* __restrict__ rpb) {
  int i = blockIdx.x * blockDim.x + threadIdx.x;
  if (i >= HEADS * NPAD * NPAD) return;
  int h = i >> 12, r = (i >> 6) & 63, c = i & 63;
  float v = 0.f;
  if (r < NTOK && c < NTOK) {
    int ri = r / 7, ci = r % 7, rj = c / 7, cj = c % 7;
    int idx = (ri - rj + 6) * 13 + (ci - cj + 6);
    v = table[idx * HEADS + h];
  }
  rpb[i] = v;
}

// ---------------- fused window attention ----------------
// one block per window; 256 threads = 8 waves; wave h handles head h
__global__ __launch_bounds__(256) void k_attn(
    const float* __restrict__ x, const float* __restrict__ n1w,
    const float* __restrict__ n1b, const float* __restrict__ qkvb,
    const float* __restrict__ projb, const fp16_t* __restrict__ wqkv,
    const fp16_t* __restrict__ wproj, const float* __restrict__ rpb,
    float* __restrict__ y) {
  extern __shared__ char smem[];
  fp16_t* xln  = (fp16_t*)smem;        // [64][XS]  (reused as attention output)
  fp16_t* qkv  = xln + NPAD * XS;      // [64][QS]
  fp16_t* pbuf = qkv + NPAD * QS;      // [8][16][PS]
  int*    gid  = (int*)(pbuf + 8 * 16 * PS);  // [64]

  const int tid = threadIdx.x, lane = tid & 31, wv = tid >> 5;
  const int wi = blockIdx.x;
  const int bb = wi >> 8, wid = wi & 255, wr = wid >> 4, wc = wid & 15;
  const int colL = lane & 15;
  const int rowHi = (lane >> 4) << 3;

  // shift-mask region id per token (Swin img_mask groups, on the fly)
  if (tid < NPAD) {
    int t = tid, tr = t / 7, tc = t % 7;
    int g = -1;
    if (t < NTOK) {
      int gr = (wr == NWSIDE - 1) ? (tr >= 4 ? 2 : 1) : 0;
      int gc = (wc == NWSIDE - 1) ? (tc >= 4 ? 2 : 1) : 0;
      g = gr * 3 + gc;
    }
    gid[t] = g;
  }

  // LayerNorm1 + shifted-window gather -> xln (f16)
  for (int t = wv; t < NPAD; t += 8) {
    if (t < NTOK) {
      int tr = t / 7, tc = t % 7;
      int rr = (wr * 7 + tr + 3) % IMG;   // roll(-3) gather
      int cc = (wc * 7 + tc + 3) % IMG;
      const float* row = x + ((size_t)(bb * IMG + rr) * IMG + cc) * DIM;
      float xv[8], s = 0.f, s2 = 0.f;
#pragma unroll
      for (int j = 0; j < 8; ++j) {
        float v = row[lane + 32 * j];
        xv[j] = v; s += v; s2 += v * v;
      }
#pragma unroll
      for (int off = 1; off < 32; off <<= 1) {
        s  += __shfl_xor(s, off, 32);
        s2 += __shfl_xor(s2, off, 32);
      }
      float mu = s * (1.f / 256.f);
      float rstd = rsqrtf(s2 * (1.f / 256.f) - mu * mu + 1e-5f);
#pragma unroll
      for (int j = 0; j < 8; ++j) {
        int ch = lane + 32 * j;
        xln[t * XS + ch] = (fp16_t)((xv[j] - mu) * rstd * n1w[ch] + n1b[ch]);
      }
    } else {
#pragma unroll
      for (int j = 0; j < 8; ++j) xln[t * XS + lane + 32 * j] = (fp16_t)0.f;
    }
  }
  __syncthreads();

  // QKV GEMM: [64,256] @ [256,768] ; 4x48 tiles over 8 waves
  for (int i = 0; i < 24; ++i) {
    int tt = wv * 24 + i;
    int m0 = (tt & 3) * 16, n0 = (tt >> 2) * 16;
    v8f c = zero8();
#pragma unroll
    for (int k0 = 0; k0 < DIM; k0 += 32) {
      v16h a = load_A(xln, XS, m0, k0);
      v16h b = load_B_kc(wqkv, DIM, k0, n0);
      c = wmma32(a, b, c);
    }
    int col = n0 + colL;
    float bias = qkvb[col];
    float sc = (col < DIM) ? 0.1767766952966369f : 1.f;  // q * hd^-0.5
#pragma unroll
    for (int v = 0; v < 8; ++v)
      qkv[(m0 + rowHi + v) * QS + col] = (fp16_t)((c[v] + bias) * sc);
  }
  __syncthreads();

  // per-head attention: wave wv = head
  const int h = wv;
  const int qoff = h * HD, koff = DIM + h * HD, voff = 2 * DIM + h * HD;
  fp16_t* pb = pbuf + h * 16 * PS;

  for (int m0 = 0; m0 < NPAD; m0 += 16) {
    v16h aq = load_A(qkv, QS, m0, qoff);
    v8f s[4];
#pragma unroll
    for (int nt = 0; nt < 4; ++nt) {
      v16h bk = load_B_kc(qkv + koff, QS, 0, nt * 16);  // K^T via row-major K
      s[nt] = wmma32(aq, bk, zero8());
    }
    // + relative position bias, + shift mask, kill padded cols
#pragma unroll
    for (int nt = 0; nt < 4; ++nt) {
      int cn = nt * 16 + colL;
#pragma unroll
      for (int v = 0; v < 8; ++v) {
        int rn = m0 + rowHi + v;
        float val = s[nt][v] + rpb[(h << 12) + (rn << 6) + cn];
        if (cn >= NTOK) val = -1e30f;
        else if (gid[rn] != gid[cn]) val += -100.f;
        s[nt][v] = val;
      }
    }
    // softmax per row (row = vgpr + half; 16 lanes hold 16 cols per tile)
#pragma unroll
    for (int v = 0; v < 8; ++v) {
      float mx = fmaxf(fmaxf(s[0][v], s[1][v]), fmaxf(s[2][v], s[3][v]));
#pragma unroll
      for (int off = 1; off < 16; off <<= 1) mx = fmaxf(mx, __shfl_xor(mx, off, 32));
      float e0 = __expf(s[0][v] - mx), e1 = __expf(s[1][v] - mx);
      float e2 = __expf(s[2][v] - mx), e3 = __expf(s[3][v] - mx);
      float sm = e0 + e1 + e2 + e3;
#pragma unroll
      for (int off = 1; off < 16; off <<= 1) sm += __shfl_xor(sm, off, 32);
      float r = 1.f / sm;
      s[0][v] = e0 * r; s[1][v] = e1 * r; s[2][v] = e2 * r; s[3][v] = e3 * r;
    }
    // stage probs f16 for A-fragment reload
#pragma unroll
    for (int nt = 0; nt < 4; ++nt)
#pragma unroll
      for (int v = 0; v < 8; ++v)
        pb[(rowHi + v) * PS + nt * 16 + colL] = (fp16_t)s[nt][v];
    __syncthreads();

    // P[16,64] @ V[64,32]
    v8f o0 = zero8(), o1 = zero8();
#pragma unroll
    for (int kt = 0; kt < 64; kt += 32) {
      v16h ap = load_A(pb, PS, 0, kt);
      v16h b0 = load_B_rm(qkv + voff, QS, kt, 0);
      v16h b1 = load_B_rm(qkv + voff, QS, kt, 16);
      o0 = wmma32(ap, b0, o0);
      o1 = wmma32(ap, b1, o1);
    }
#pragma unroll
    for (int v = 0; v < 8; ++v) {
      int rn = m0 + rowHi + v;
      xln[rn * XS + h * HD + colL]      = (fp16_t)o0[v];   // xln reused as out
      xln[rn * XS + h * HD + 16 + colL] = (fp16_t)o1[v];
    }
    __syncthreads();
  }

  // proj GEMM + residual, scatter back (roll +3 == same source index)
  for (int i = 0; i < 8; ++i) {
    int tt = wv * 8 + i;
    int m0 = (tt >> 4) * 16, n0 = (tt & 15) * 16;
    v8f c = zero8();
#pragma unroll
    for (int k0 = 0; k0 < DIM; k0 += 32) {
      v16h a = load_A(xln, XS, m0, k0);
      v16h b = load_B_kc(wproj, DIM, k0, n0);
      c = wmma32(a, b, c);
    }
    int col = n0 + colL;
    float bias = projb[col];
#pragma unroll
    for (int v = 0; v < 8; ++v) {
      int t = m0 + rowHi + v;
      if (t < NTOK) {
        int tr = t / 7, tc = t % 7;
        int rr = (wr * 7 + tr + 3) % IMG;
        int cc = (wc * 7 + tc + 3) % IMG;
        size_t gi = ((size_t)(bb * IMG + rr) * IMG + cc) * DIM + col;
        y[gi] = x[gi] + c[v] + bias;
      }
    }
  }
}

// ---------------- fused MLP (LN2 -> fc1 -> GELU -> fc2 -> +res), in-place on y ----
__global__ __launch_bounds__(256) void k_mlp(
    const float* __restrict__ n2w, const float* __restrict__ n2b,
    const float* __restrict__ b1, const float* __restrict__ b2,
    const fp16_t* __restrict__ w1, const fp16_t* __restrict__ w2,
    float* __restrict__ y) {
  extern __shared__ char smem[];
  fp16_t* a  = (fp16_t*)smem;      // [64][XS] LN'd activations
  fp16_t* hb = a + NPAD * XS;      // [64][XS] hidden chunk (GELU'd)

  const int tid = threadIdx.x, lane = tid & 31, wv = tid >> 5;
  const int colL = lane & 15;
  const int rowHi = (lane >> 4) << 3;
  const size_t base = (size_t)blockIdx.x * NPAD;  // 3136 * 64 = all tokens

  // LayerNorm2 of y1 rows -> a (f16)
  for (int t = wv; t < NPAD; t += 8) {
    const float* row = y + (base + t) * DIM;
    float xv[8], s = 0.f, s2 = 0.f;
#pragma unroll
    for (int j = 0; j < 8; ++j) {
      float v = row[lane + 32 * j];
      xv[j] = v; s += v; s2 += v * v;
    }
#pragma unroll
    for (int off = 1; off < 32; off <<= 1) {
      s  += __shfl_xor(s, off, 32);
      s2 += __shfl_xor(s2, off, 32);
    }
    float mu = s * (1.f / 256.f);
    float rstd = rsqrtf(s2 * (1.f / 256.f) - mu * mu + 1e-5f);
#pragma unroll
    for (int j = 0; j < 8; ++j) {
      int ch = lane + 32 * j;
      a[t * XS + ch] = (fp16_t)((xv[j] - mu) * rstd * n2w[ch] + n2b[ch]);
    }
  }
  __syncthreads();

  v8f acc[8];
#pragma unroll
  for (int i = 0; i < 8; ++i) acc[i] = zero8();

  for (int chunk = 0; chunk < 4; ++chunk) {
    // fc1 chunk: [64,256] @ [256,256] -> GELU -> hb
    for (int i = 0; i < 8; ++i) {
      int tt = wv * 8 + i;
      int m0 = (tt >> 4) * 16, n0 = (tt & 15) * 16;
      v8f c = zero8();
#pragma unroll
      for (int k0 = 0; k0 < DIM; k0 += 32) {
        v16h af = load_A(a, XS, m0, k0);
        v16h bf = load_B_kc(w1 + (size_t)chunk * 256 * DIM, DIM, k0, n0);
        c = wmma32(af, bf, c);
      }
      int col = n0 + colL;
      float bias = b1[chunk * 256 + col];
#pragma unroll
      for (int v = 0; v < 8; ++v) {
        float u = c[v] + bias;
        u = 0.5f * u * (1.f + erff(u * 0.70710678118654752f));  // exact GELU
        hb[(m0 + rowHi + v) * XS + col] = (fp16_t)u;
      }
    }
    __syncthreads();
    // fc2 partial: hb[64,256] @ w2_chunk[256,256], accumulate in registers
    for (int i = 0; i < 8; ++i) {
      int tt = wv * 8 + i;
      int m0 = (tt >> 4) * 16, n0 = (tt & 15) * 16;
#pragma unroll
      for (int k0 = 0; k0 < DIM; k0 += 32) {
        v16h af = load_A(hb, XS, m0, k0);
        v16h bf = load_B_kc(w2, HID, chunk * 256 + k0, n0);
        acc[i] = wmma32(af, bf, acc[i]);
      }
    }
    __syncthreads();
  }

  // epilogue: y = y1 + fc2 + b2  (each element read+written by its unique owner)
  for (int i = 0; i < 8; ++i) {
    int tt = wv * 8 + i;
    int m0 = (tt >> 4) * 16, n0 = (tt & 15) * 16;
    int col = n0 + colL;
    float bias = b2[col];
#pragma unroll
    for (int v = 0; v < 8; ++v) {
      size_t gi = (base + m0 + rowHi + v) * DIM + col;
      y[gi] = y[gi] + acc[i][v] + bias;
    }
  }
}

// ---------------- launch ----------------
extern "C" void kernel_launch(void* const* d_in, const int* in_sizes, int n_in,
                              void* d_out, int out_size, void* d_ws, size_t ws_size,
                              hipStream_t stream) {
  const float* x     = (const float*)d_in[0];
  const float* n1w   = (const float*)d_in[1];
  const float* n1b   = (const float*)d_in[2];
  const float* qkvw  = (const float*)d_in[3];
  const float* qkvb  = (const float*)d_in[4];
  const float* projw = (const float*)d_in[5];
  const float* projb = (const float*)d_in[6];
  const float* btab  = (const float*)d_in[7];
  const float* n2w   = (const float*)d_in[8];
  const float* n2b   = (const float*)d_in[9];
  const float* fc1w  = (const float*)d_in[10];
  const float* fc1b  = (const float*)d_in[11];
  const float* fc2w  = (const float*)d_in[12];
  const float* fc2b  = (const float*)d_in[13];
  float* y = (float*)d_out;

  char* ws = (char*)d_ws;
  fp16_t* wqkv_t  = (fp16_t*)(ws);                 // [768][256]  384 KB
  fp16_t* wproj_t = (fp16_t*)(ws + 393216);        // [256][256]  128 KB
  fp16_t* wfc1_t  = (fp16_t*)(ws + 524288);        // [1024][256] 512 KB
  fp16_t* wfc2_t  = (fp16_t*)(ws + 1048576);       // [256][1024] 512 KB
  float*  rpb     = (float*)(ws + 1572864);        // [8][64][64] 128 KB

  k_transpose_f16<<<(256 * 768 + 255) / 256, 256, 0, stream>>>(qkvw, wqkv_t, 256, 768);
  k_transpose_f16<<<(256 * 256 + 255) / 256, 256, 0, stream>>>(projw, wproj_t, 256, 256);
  k_transpose_f16<<<(256 * 1024 + 255) / 256, 256, 0, stream>>>(fc1w, wfc1_t, 256, 1024);
  k_transpose_f16<<<(1024 * 256 + 255) / 256, 256, 0, stream>>>(fc2w, wfc2_t, 1024, 256);
  k_rpb<<<(HEADS * NPAD * NPAD + 255) / 256, 256, 0, stream>>>(btab, rpb);

  size_t smem_attn = (size_t)(NPAD * XS + NPAD * QS + 8 * 16 * PS) * sizeof(fp16_t)
                   + NPAD * sizeof(int);
  k_attn<<<4096, 256, smem_attn, stream>>>(x, n1w, n1b, qkvb, projb,
                                           wqkv_t, wproj_t, rpb, y);

  size_t smem_mlp = (size_t)(2 * NPAD * XS) * sizeof(fp16_t);
  k_mlp<<<3136, 256, smem_mlp, stream>>>(n2w, n2b, fc1b, fc2b, wfc1_t, wfc2_t, y);

  (void)in_sizes; (void)n_in; (void)out_size; (void)ws_size;
}
